// _BaseTreeBackbone_45449343926446
// MI455X (gfx1250) — compile-verified, matrix operand order checked by reading, not run on hardware
//
#include <hip/hip_runtime.h>

typedef __attribute__((ext_vector_type(16))) __bf16 v16bf;
typedef __attribute__((ext_vector_type(8)))  __bf16 v8bf;
typedef __attribute__((ext_vector_type(8)))  float  v8f;

#define MM 256      // rows of A / cols of residual
#define NN 1024     // cols of A / cols of x
#define BB 16384    // batch
#define KITER 16    // ALISTA iterations

// Load a 16-element bf16 WMMA fragment for this lane.
// ISA layout (16-bit A/B 16x32 / 32x16): lane holds row (lane&15); lanes<16 own
// K = 0..7 and 16..23, lanes>=16 own K = 8..15 and 24..31.
// Caller passes the row pointer already offset by khalf = ((lane>>4)<<3).
__device__ __forceinline__ v16bf load_frag(const __bf16* __restrict__ p) {
  v8bf lo = *(const v8bf*)(p);
  v8bf hi = *(const v8bf*)(p + 16);
  v16bf f;
#pragma unroll
  for (int i = 0; i < 8; ++i) { f[i] = lo[i]; f[i + 8] = hi[i]; }
  return f;
}

__device__ __forceinline__ v8f wmma_bf16(v16bf a, v16bf b, v8f c) {
  return __builtin_amdgcn_wmma_f32_16x16x32_bf16(false, a, false, b,
                                                 (short)0, c, false, false);
}

// One-time prep: Ab[m][n] = bf16(A[m][n]);  Wt[n][m] = bf16(W[m][n])
__global__ __launch_bounds__(256) void alista_prep(
    const float* __restrict__ A, const float* __restrict__ W,
    __bf16* __restrict__ Ab, __bf16* __restrict__ Wt) {
  int idx = blockIdx.x * 256 + threadIdx.x;        // 0 .. 256*1024-1
  int m = idx >> 10;
  int n = idx & (NN - 1);
  Ab[idx] = (__bf16)A[idx];
  Wt[(size_t)n * MM + m] = (__bf16)W[idx];
}

// residual: rb[b][m] = bf16( y[b][m] - sum_n xb[b][n] * Ab[m][n] )
// Block tile 128(b) x 128(m); 8 waves as 4x2; wave tile 32(b) x 64(m):
// per k-step 6 fragment loads -> 8 WMMAs (1.5 b128 loads per WMMA).
__global__ __launch_bounds__(256) void alista_resid(
    const float* __restrict__ y, const __bf16* __restrict__ xb,
    const __bf16* __restrict__ Ab, __bf16* __restrict__ rb) {
  const int lane  = threadIdx.x & 31;
  const int wave  = threadIdx.x >> 5;
  const int b0    = blockIdx.y * 128 + (wave >> 1) * 32;
  const int m0    = blockIdx.x * 128 + (wave & 1) * 64;
  const int lr    = lane & 15;
  const int khalf = (lane >> 4) << 3;              // 0 or 8

  v8f acc[2][4];
#pragma unroll
  for (int t = 0; t < 2; ++t)
#pragma unroll
    for (int s = 0; s < 4; ++s) acc[t][s] = (v8f){};

  const __bf16* xrow0 = xb + (size_t)(b0 + lr) * NN + khalf;
  const __bf16* xrow1 = xrow0 + (size_t)16 * NN;
  const __bf16* arow0 = Ab + (size_t)(m0 + lr) * NN + khalf;

  for (int kk = 0; kk < NN; kk += 32) {
    v16bf a0 = load_frag(xrow0 + kk);
    v16bf a1 = load_frag(xrow1 + kk);
#pragma unroll
    for (int s = 0; s < 4; ++s) {
      v16bf bf = load_frag(arow0 + (size_t)(s * 16) * NN + kk);
      acc[0][s] = wmma_bf16(a0, bf, acc[0][s]);
      acc[1][s] = wmma_bf16(a1, bf, acc[1][s]);
    }
  }

  // C/D layout: VGPR j -> row base + j + (lane>=16 ? 8 : 0); col = lane&15.
#pragma unroll
  for (int t = 0; t < 2; ++t) {
    const int rowb = b0 + t * 16 + ((lane >> 4) << 3);
#pragma unroll
    for (int s = 0; s < 4; ++s) {
      const int c = m0 + s * 16 + lr;
#pragma unroll
      for (int j = 0; j < 8; ++j) {
        size_t idx = (size_t)(rowb + j) * MM + c;
        rb[idx] = (__bf16)(y[idx] - acc[t][s][j]);
      }
    }
  }
}

// update: u = x[b][n] + gamma * sum_m rb[b][m] * Wt[n][m]
//         xnew = soft(u, theta); write fp32 x (in place) and bf16 shadow xb.
// Block tile 128(b) x 128(n); wave tile 32(b) x 64(n); K = MM step 32.
__global__ __launch_bounds__(256) void alista_update(
    const __bf16* __restrict__ rb, const __bf16* __restrict__ Wt,
    float* __restrict__ x, __bf16* __restrict__ xb,
    const float* __restrict__ steps, const float* __restrict__ thrs, int k) {
  const int lane  = threadIdx.x & 31;
  const int wave  = threadIdx.x >> 5;
  const int b0    = blockIdx.y * 128 + (wave >> 1) * 32;
  const int n0    = blockIdx.x * 128 + (wave & 1) * 64;
  const int lr    = lane & 15;
  const int khalf = (lane >> 4) << 3;

  v8f acc[2][4];
#pragma unroll
  for (int t = 0; t < 2; ++t)
#pragma unroll
    for (int s = 0; s < 4; ++s) acc[t][s] = (v8f){};

  const __bf16* rrow0 = rb + (size_t)(b0 + lr) * MM + khalf;
  const __bf16* rrow1 = rrow0 + (size_t)16 * MM;
  const __bf16* wrow0 = Wt + (size_t)(n0 + lr) * MM + khalf;

#pragma unroll
  for (int kk = 0; kk < MM; kk += 32) {
    v16bf a0 = load_frag(rrow0 + kk);
    v16bf a1 = load_frag(rrow1 + kk);
#pragma unroll
    for (int s = 0; s < 4; ++s) {
      v16bf bf = load_frag(wrow0 + (size_t)(s * 16) * MM + kk);
      acc[0][s] = wmma_bf16(a0, bf, acc[0][s]);
      acc[1][s] = wmma_bf16(a1, bf, acc[1][s]);
    }
  }

  const float gamma = fabsf(steps[k]);
  const float theta = fabsf(thrs[k]);
#pragma unroll
  for (int t = 0; t < 2; ++t) {
    const int rowb = b0 + t * 16 + ((lane >> 4) << 3);
#pragma unroll
    for (int s = 0; s < 4; ++s) {
      const int c = n0 + s * 16 + lr;
#pragma unroll
      for (int j = 0; j < 8; ++j) {
        size_t idx = (size_t)(rowb + j) * NN + c;
        float u = x[idx] + gamma * acc[t][s][j];
        float m = fabsf(u) - theta;
        m = m > 0.f ? m : 0.f;
        float o = (u >= 0.f) ? m : -m;
        x[idx]  = o;
        xb[idx] = (__bf16)o;
      }
    }
  }
}

extern "C" void kernel_launch(void* const* d_in, const int* in_sizes, int n_in,
                              void* d_out, int out_size, void* d_ws, size_t ws_size,
                              hipStream_t stream) {
  const float* y     = (const float*)d_in[0];   // [BB, MM]
  const float* A     = (const float*)d_in[1];   // [MM, NN]
  const float* W     = (const float*)d_in[2];   // [MM, NN]
  const float* steps = (const float*)d_in[3];   // [KITER]
  const float* thrs  = (const float*)d_in[4];   // [KITER]

  float* x = (float*)d_out;                     // [BB, NN] fp32 master state

  char* ws = (char*)d_ws;
  __bf16* xb = (__bf16*)ws;                                    // 32 MB shadow of x
  __bf16* rb = (__bf16*)(ws + (size_t)BB * NN * 2);            //  8 MB residual
  __bf16* Ab = (__bf16*)(ws + (size_t)BB * NN * 2
                            + (size_t)BB * MM * 2);            // 0.5 MB
  __bf16* Wt = Ab + (size_t)MM * NN;                           // 0.5 MB (transposed)

  // x0 = 0 (both fp32 master and bf16 shadow); capture-safe memsets.
  hipMemsetAsync(x,  0, (size_t)BB * NN * 4, stream);
  hipMemsetAsync(xb, 0, (size_t)BB * NN * 2, stream);

  alista_prep<<<dim3((MM * NN) / 256), 256, 0, stream>>>(A, W, Ab, Wt);

  for (int k = 0; k < KITER; ++k) {
    alista_resid<<<dim3(MM / 128, BB / 128), 256, 0, stream>>>(y, xb, Ab, rb);
    alista_update<<<dim3(NN / 128, BB / 128), 256, 0, stream>>>(rb, Wt, x, xb,
                                                                steps, thrs, k);
  }
}
